// FlashAttention2_1692217115253
// MI455X (gfx1250) — compile-verified, matrix operand order checked by reading, not run on hardware
//
#include <hip/hip_runtime.h>
#include <cstdint>
#include <cstddef>

typedef __attribute__((ext_vector_type(16))) __bf16 v16bf;
typedef __attribute__((ext_vector_type(8)))  float  v8f;

namespace {

constexpr int kT   = 2048;   // sequence length
constexpr int kH   = 16;     // heads
constexpr int kHD  = 64;     // head dim
constexpr int kC   = 1024;   // model dim
constexpr int kWin = 511;    // attend |i-j| <= 511

// f32 -> bf16 round-to-nearest-even, pure integer ops (no bf16 VALU dependence)
__device__ __forceinline__ __bf16 f2bf(float f) {
  union { float f; uint32_t u; } x; x.f = f;
  uint32_t r = x.u + 0x7fffu + ((x.u >> 16) & 1u);
  union { uint16_t s; __bf16 b; } y; y.s = (uint16_t)(r >> 16);
  return y.b;
}

union FragU { v16bf v; uint4 u[2]; };

// Load a 16x32 bf16 fragment in the CDNA5 16-bit A/B layout:
// lane L (L<16): row M=L, K in {0..7, 16..23}; lane L+16: row M=L, K in {8..15, 24..31}.
// Per lane: two contiguous 16-byte chunks at K offsets 8h and 16+8h.
__device__ __forceinline__ v16bf load_frag(const __bf16* base, int ld) {
  const int lane = threadIdx.x & 31;
  const int r = lane & 15;
  const int h = lane >> 4;
  const __bf16* p = base + r * ld + 8 * h;
  FragU f;
  f.u[0] = *(const uint4*)(p);
  f.u[1] = *(const uint4*)(p + 16);
  return f.v;
}

__device__ __forceinline__ v8f wmma_bf(v16bf a, v16bf b, v8f c) {
  return __builtin_amdgcn_wmma_f32_16x16x32_bf16(
      /*neg_a=*/false, a, /*neg_b=*/false, b,
      /*c_mod=*/(short)0, c, /*reuse_a=*/false, /*reuse_b=*/false);
}

// 32x16 all-ones bf16 B fragment: rowsum(P) = P x ones via one WMMA.
__device__ __forceinline__ v16bf ones_frag() {
  union { v16bf v; uint32_t w[8]; } f;
#pragma unroll
  for (int i = 0; i < 8; ++i) f.w[i] = 0x3f803f80u;  // bf16 1.0 pair
  return f.v;
}

} // namespace

__global__ void __launch_bounds__(256)
cvt_f32_to_bf16(const float* __restrict__ src, __bf16* __restrict__ dst, int n) {
  int i = (blockIdx.x * 256 + threadIdx.x) * 8;
  if (i + 8 > n) return;
  float4 a = *(const float4*)(src + i);
  float4 b = *(const float4*)(src + i + 4);
  union { __bf16 h[8]; uint4 u; } o;
  o.h[0] = f2bf(a.x); o.h[1] = f2bf(a.y); o.h[2] = f2bf(a.z); o.h[3] = f2bf(a.w);
  o.h[4] = f2bf(b.x); o.h[5] = f2bf(b.y); o.h[6] = f2bf(b.z); o.h[7] = f2bf(b.w);
  *(uint4*)(dst + i) = o.u;
}

// D[m,n] = sum_k A[m,k] * B[n,k]   (A: MxK, B: NxK, both bf16 row-major)
// MODE 0: bf16 out, [B,H,T,hd] layout (Q/K)
// MODE 1: bf16 out, [B,H,hd,T] layout (V transposed)
// MODE 2: f32 out, row-major MxN (final projection)
template <int MODE>
__global__ void __launch_bounds__(256)
gemm_bt(const __bf16* __restrict__ A, const __bf16* __restrict__ Bm,
        void* __restrict__ out, int M, int N, int K) {
  const int w   = blockIdx.x * 8 + (threadIdx.x >> 5);
  const int ntn = N >> 5;
  const int m0  = (w / ntn) << 5;
  const int n0  = (w % ntn) << 5;
  (void)M;
  v8f acc[2][2] = {};
  for (int k0 = 0; k0 < K; k0 += 32) {
    v16bf a0 = load_frag(A  + (size_t)m0 * K + k0, K);
    v16bf a1 = load_frag(A  + (size_t)(m0 + 16) * K + k0, K);
    v16bf b0 = load_frag(Bm + (size_t)n0 * K + k0, K);
    v16bf b1 = load_frag(Bm + (size_t)(n0 + 16) * K + k0, K);
    acc[0][0] = wmma_bf(a0, b0, acc[0][0]);
    acc[0][1] = wmma_bf(a0, b1, acc[0][1]);
    acc[1][0] = wmma_bf(a1, b0, acc[1][0]);
    acc[1][1] = wmma_bf(a1, b1, acc[1][1]);
  }
  // C/D layout: lane<16 -> (M = v + 0, N = lane); lane>=16 -> (M = v + 8, N = lane-16)
  const int lane = threadIdx.x & 31;
  const int cc = lane & 15;
  const int hh = lane >> 4;
#pragma unroll
  for (int i = 0; i < 2; ++i)
#pragma unroll
    for (int j = 0; j < 2; ++j)
#pragma unroll
      for (int v = 0; v < 8; ++v) {
        int m = m0 + 16 * i + v + 8 * hh;
        int n = n0 + 16 * j + cc;
        float val = acc[i][j][v];
        if (MODE == 2) {
          ((float*)out)[(size_t)m * N + n] = val;
        } else {
          int b = m >> 11, t = m & (kT - 1);
          int head = n >> 6, d = n & (kHD - 1);
          __bf16 bv = f2bf(val);
          if (MODE == 0)
            ((__bf16*)out)[(((size_t)(b * kH + head) * kT) + t) * kHD + d] = bv;
          else
            ((__bf16*)out)[((size_t)(b * kH + head) * kHD + d) * kT + t] = bv;
        }
      }
}

// One wave per 16-row query tile. Q:[BH,T,hd] bf16, K:[BH,T,hd] bf16, V:[BH,hd,T] bf16.
// Output aout: [B*T, C] bf16.
__global__ void __launch_bounds__(256)
flash_swa(const __bf16* __restrict__ Q, const __bf16* __restrict__ Kt,
          const __bf16* __restrict__ Vt, __bf16* __restrict__ aout) {
  __shared__ __align__(16) __bf16 pbuf[8][16 * 32];
  const int wave = threadIdx.x >> 5;
  const int lane = threadIdx.x & 31;
  const int cc = lane & 15;
  const int hh = lane >> 4;
  const int g  = blockIdx.x * 8 + wave;
  const int bh = g >> 7;               // 128 query tiles per (b,h)
  const int q0 = (g & 127) << 4;

  const __bf16* Qp = Q  + (size_t)bh * kT * kHD;
  const __bf16* Kp = Kt + (size_t)bh * kT * kHD;
  const __bf16* Vp = Vt + (size_t)bh * kHD * kT;

  const v16bf qa0 = load_frag(Qp + q0 * kHD, kHD);
  const v16bf qa1 = load_frag(Qp + q0 * kHD + 32, kHD);
  const v16bf ones = ones_frag();

  v8f o0 = {}, o1 = {}, o2 = {}, o3 = {};
  float mrow[8], lrow[8];
#pragma unroll
  for (int v = 0; v < 8; ++v) { mrow[v] = -3.0e38f; lrow[v] = 0.f; }
  const float ce = 0.18033688011112042f;  // (1/sqrt(hd)) * log2(e)

  int jstart = (q0 - kWin) & ~31; if (jstart < 0) jstart = 0;
  int jend = q0 + 15 + kWin;      if (jend > kT - 1) jend = kT - 1;

  for (int jt = jstart; jt <= jend; jt += 32) {
    // Prefetch next key tile (speculative; dropped if translation fails).
    __builtin_prefetch(Kp + (size_t)(jt + 32) * kHD, 0, 1);
    __builtin_prefetch(Vp + jt + 32, 0, 1);

    v16bf k00 = load_frag(Kp + (size_t)jt * kHD, kHD);
    v16bf k01 = load_frag(Kp + (size_t)jt * kHD + 32, kHD);
    v16bf k10 = load_frag(Kp + (size_t)(jt + 16) * kHD, kHD);
    v16bf k11 = load_frag(Kp + (size_t)(jt + 16) * kHD + 32, kHD);
    v8f s0 = {}, s1 = {};
    s0 = wmma_bf(qa0, k00, s0);
    s0 = wmma_bf(qa1, k01, s0);
    s1 = wmma_bf(qa0, k10, s1);
    s1 = wmma_bf(qa1, k11, s1);

    // Tile fully inside the window for every row?  (skip masking if so)
    const bool edge = (jt < q0 - (kWin - 15)) || (jt > q0 + (kWin - 31));
    float alpha[8], p0[8], p1[8];
#pragma unroll
    for (int v = 0; v < 8; ++v) {
      const int i = q0 + v + 8 * hh;
      float sv0 = s0[v], sv1 = s1[v];
      if (edge) {
        int dj0 = i - (jt + cc);      if (dj0 < 0) dj0 = -dj0;
        int dj1 = i - (jt + 16 + cc); if (dj1 < 0) dj1 = -dj1;
        if (dj0 > kWin) sv0 = -3.0e38f;
        if (dj1 > kWin) sv1 = -3.0e38f;
      }
      float mt = fmaxf(sv0, sv1);
#pragma unroll
      for (int off = 1; off < 16; off <<= 1) mt = fmaxf(mt, __shfl_xor(mt, off, 32));
      const float mn = fmaxf(mrow[v], mt);
      const float a  = exp2f((mrow[v] - mn) * ce);
      float e0 = exp2f((sv0 - mn) * ce);
      float e1 = exp2f((sv1 - mn) * ce);
      if (sv0 < -1.0e37f) e0 = 0.f;   // fully-masked guard
      if (sv1 < -1.0e37f) e1 = 0.f;
      mrow[v] = mn;
      alpha[v] = a;
      p0[v] = e0;
      p1[v] = e1;
    }
#pragma unroll
    for (int v = 0; v < 8; ++v) {
      o0[v] *= alpha[v]; o1[v] *= alpha[v]; o2[v] *= alpha[v]; o3[v] *= alpha[v];
    }
    // C-layout -> A-layout transpose of P via per-wave LDS slice (DS ops are
    // in-order within a wave; compiler inserts the dscnt wait for the reload).
    __bf16* pb = pbuf[wave];
#pragma unroll
    for (int v = 0; v < 8; ++v) {
      pb[(v + 8 * hh) * 32 + cc]      = f2bf(p0[v]);
      pb[(v + 8 * hh) * 32 + 16 + cc] = f2bf(p1[v]);
    }
    const v16bf pa = load_frag(pb, 32);

    // Row sums of P on the XDL pipe: P (16x32) x ones (32x16) -> every column
    // of the C fragment holds the row's sum; replaces a 4-level shuffle tree.
    v8f zero = {};
    v8f srow = wmma_bf(pa, ones, zero);
#pragma unroll
    for (int v = 0; v < 8; ++v) lrow[v] = lrow[v] * alpha[v] + srow[v];

    v16bf vb0 = load_frag(Vp + jt, kT);              // rows d=0..15, K=j contiguous
    v16bf vb1 = load_frag(Vp + 16 * kT + jt, kT);
    v16bf vb2 = load_frag(Vp + 32 * kT + jt, kT);
    v16bf vb3 = load_frag(Vp + 48 * kT + jt, kT);
    o0 = wmma_bf(pa, vb0, o0);
    o1 = wmma_bf(pa, vb1, o1);
    o2 = wmma_bf(pa, vb2, o2);
    o3 = wmma_bf(pa, vb3, o3);
  }

  const int b = bh >> 4, head = bh & 15;
#pragma unroll
  for (int v = 0; v < 8; ++v) {
    const float inv = 1.0f / lrow[v];
    const int t = q0 + v + 8 * hh;
    __bf16* dst = aout + ((size_t)(b * kT + t)) * kC + head * kHD;
    dst[cc]      = f2bf(o0[v] * inv);
    dst[16 + cc] = f2bf(o1[v] * inv);
    dst[32 + cc] = f2bf(o2[v] * inv);
    dst[48 + cc] = f2bf(o3[v] * inv);
  }
}

extern "C" void kernel_launch(void* const* d_in, const int* in_sizes, int n_in,
                              void* d_out, int out_size, void* d_ws, size_t ws_size,
                              hipStream_t stream) {
  (void)in_sizes; (void)n_in; (void)out_size; (void)ws_size;
  const float* x  = (const float*)d_in[0];
  const float* Wq = (const float*)d_in[1];
  const float* Wk = (const float*)d_in[2];
  const float* Wv = (const float*)d_in[3];
  const float* Wo = (const float*)d_in[4];

  char* ws = (char*)d_ws;
  const size_t MB = 1024ull * 1024ull;
  __bf16* xb  = (__bf16*)(ws + 0 * MB);   // [4096,1024]  8 MB
  __bf16* wqb = (__bf16*)(ws + 8 * MB);   // [1024,1024]  2 MB
  __bf16* wkb = (__bf16*)(ws + 10 * MB);
  __bf16* wvb = (__bf16*)(ws + 12 * MB);
  __bf16* wob = (__bf16*)(ws + 14 * MB);
  __bf16* Qb  = (__bf16*)(ws + 16 * MB);  // [32,2048,64] 8 MB
  __bf16* Kb  = (__bf16*)(ws + 24 * MB);  // [32,2048,64] 8 MB
  __bf16* Vtb = (__bf16*)(ws + 32 * MB);  // [32,64,2048] 8 MB (transposed)
  __bf16* ab  = (__bf16*)(ws + 40 * MB);  // [4096,1024]  8 MB

  // Phase 0: bf16 conversion
  cvt_f32_to_bf16<<<2048, 256, 0, stream>>>(x,  xb,  2 * 2048 * 1024);
  cvt_f32_to_bf16<<<512,  256, 0, stream>>>(Wq, wqb, 1024 * 1024);
  cvt_f32_to_bf16<<<512,  256, 0, stream>>>(Wk, wkb, 1024 * 1024);
  cvt_f32_to_bf16<<<512,  256, 0, stream>>>(Wv, wvb, 1024 * 1024);
  cvt_f32_to_bf16<<<512,  256, 0, stream>>>(Wo, wob, 1024 * 1024);

  // Phase 1: QKV projections.  (4096/32)*(1024/32)=4096 waves -> 512 blocks of 8 waves.
  gemm_bt<0><<<512, 256, 0, stream>>>(xb, wqb, Qb,  4096, 1024, 1024);
  gemm_bt<0><<<512, 256, 0, stream>>>(xb, wkb, Kb,  4096, 1024, 1024);
  gemm_bt<1><<<512, 256, 0, stream>>>(xb, wvb, Vtb, 4096, 1024, 1024);

  // Phase 2: sliding-window flash attention. 32 bh * 128 qtiles = 4096 waves.
  flash_swa<<<512, 256, 0, stream>>>(Qb, Kb, Vtb, ab);

  // Phase 3: output projection, f32 result straight to d_out.
  gemm_bt<2><<<512, 256, 0, stream>>>(ab, wob, d_out, 4096, 1024, 1024);
}
